// MGMQTorchModel_13898514170006
// MI455X (gfx1250) — matrix operand order repaired and checked by reading, not compiled, not run on hardware
//
#include <hip/hip_runtime.h>
#include <hip/hip_bf16.h>
#include <math.h>

typedef __attribute__((ext_vector_type(16))) _Float16 v16h;
typedef __attribute__((ext_vector_type(8)))  _Float16 v8h;
typedef __attribute__((ext_vector_type(8)))  float    v8f;

#define B_SZ   8192
#define KNBR   4
#define NODES  (B_SZ * (KNBR + 1))   // 40960
#define EMBSZ  128
#define AST    264                   // kernel1 Afrag row stride (f16): conflict-free ds_load_b128

// f16 transposed-weight arena layout (element offsets in _Float16 units)
#define WFT_OFF   0                          // wfuseT  [128][256]
#define WPT_OFF   (128 * 256)                // WpT     [64][160]  (K padded 129->160 w/ zeros)
#define WIHF_OFF  (WPT_OFF  + 64 * 160)      // WihF^T  [96][64]
#define WIHB_OFF  (WIHF_OFF + 96 * 64)       // WihB^T  [96][64]
#define WHHF_OFF  (WIHB_OFF + 96 * 64)       // WhhF^T  [96][32]
#define WHHB_OFF  (WHHF_OFF + 96 * 32)       // WhhB^T  [96][32]
#define WOT_OFF   (WHHB_OFF + 96 * 32)       // Wo^T    [64][192]
#define PREP_TOTAL (WOT_OFF + 64 * 192)      // 73728 f16 = 147456 B

// head-kernel LDS strides (f16), padded for conflict-free b128 row access
#define ABST 200   // Abuf: 16 x 200  (proj A 160 cols / final A 192 cols)
#define XST  72    // xbuf: 16 x 72   (x_t, 64 cols)
#define HST  40    // hbuf: 16 x 40   (h, 32 cols)

__device__ __forceinline__ float elu_f(float x)  { return x > 0.f ? x : (__expf(x) - 1.f); }
__device__ __forceinline__ float sigm_f(float x) { return 1.f / (1.f + __expf(-x)); }

__device__ __forceinline__ v8f wmma_f16(v16h a, v16h b, v8f c) {
    return __builtin_amdgcn_wmma_f32_16x16x32_f16(false, a, false, b, (short)0, c, false, false);
}
// A-fragment (16x32 f16): lane half selects 16 contiguous K [koff..koff+15]
__device__ __forceinline__ v16h load_a_frag(const _Float16* base) {
    v8h lo = *(const v8h*)base;
    v8h hi = *(const v8h*)(base + 8);
    v16h a;
    #pragma unroll
    for (int q = 0; q < 8; q++) { a[q] = lo[q]; a[q + 8] = hi[q]; }
    return a;
}
// B-fragment (32x16 f16, n-major source): 16 contiguous K per lane-half
__device__ __forceinline__ v16h load_b_frag(const _Float16* p) { return *(const v16h*)p; }

// ---------------------------------------------------------------------------
// Kernel 0: transpose + f16-convert all GEMM weights into the ws arena.
// ---------------------------------------------------------------------------
__global__ void prep_weights(const float* __restrict__ Wf,   const float* __restrict__ Wp,
                             const float* __restrict__ WihF, const float* __restrict__ WihB,
                             const float* __restrict__ WhhF, const float* __restrict__ WhhB,
                             const float* __restrict__ Wo,   _Float16* __restrict__ wt)
{
    int idx = blockIdx.x * blockDim.x + threadIdx.x;
    if (idx < 128 * 256) { int n = idx >> 8, k = idx & 255;
        wt[WFT_OFF + n * 256 + k] = (_Float16)Wf[k * 128 + n]; return; }
    idx -= 128 * 256;
    if (idx < 64 * 160) { int n = idx / 160, k = idx % 160;
        wt[WPT_OFF + n * 160 + k] = (_Float16)(k < 129 ? Wp[k * 64 + n] : 0.f); return; }
    idx -= 64 * 160;
    if (idx < 96 * 64)  { int g = idx >> 6, c = idx & 63;
        wt[WIHF_OFF + g * 64 + c] = (_Float16)WihF[c * 96 + g]; return; }
    idx -= 96 * 64;
    if (idx < 96 * 64)  { int g = idx >> 6, c = idx & 63;
        wt[WIHB_OFF + g * 64 + c] = (_Float16)WihB[c * 96 + g]; return; }
    idx -= 96 * 64;
    if (idx < 96 * 32)  { int g = idx >> 5, c = idx & 31;
        wt[WHHF_OFF + g * 32 + c] = (_Float16)WhhF[c * 96 + g]; return; }
    idx -= 96 * 32;
    if (idx < 96 * 32)  { int g = idx >> 5, c = idx & 31;
        wt[WHHB_OFF + g * 32 + c] = (_Float16)WhhB[c * 96 + g]; return; }
    idx -= 96 * 32;
    if (idx < 64 * 192) { int n = idx / 192, c = idx % 192;
        wt[WOT_OFF + n * 192 + c] = (_Float16)Wo[c * 64 + n]; return; }
}

// ---------------------------------------------------------------------------
// Kernel 1: GAT (coop+conf) + fused GEMM (WMMA) + lane-mean. One wave = node.
//   out = att @ (x@W) rewritten as (att@x) @ W  -> ~2x VALU reduction.
// ---------------------------------------------------------------------------
__global__ __launch_bounds__(128) void gat_fuse_kernel(
    const float* __restrict__ self_f,  const float* __restrict__ nbr_f,
    const float* __restrict__ W_coop,  const float* __restrict__ as_coop, const float* __restrict__ ad_coop,
    const float* __restrict__ W_conf,  const float* __restrict__ as_conf, const float* __restrict__ ad_conf,
    const _Float16* __restrict__ wfT,  const float* __restrict__ b_fuse,
    float* __restrict__ emb_out)
{
    __shared__ float wssh[2][4][4];
    __shared__ float wdsh[2][4][4];
    __shared__ float xsh[4][48];
    __shared__ float ysh[4][12][4];                       // y = att @ x  per wave
    __shared__ __align__(16) _Float16 Afrag[4][16][AST];  // fuse-GEMM A (12 rows + 4 zero pad)

    const int tid  = threadIdx.x;
    const int w    = tid >> 5;
    const int lane = tid & 31;

    // ws/wd = W[h] @ a_{src,dst}[h] : 64 threads, one dot-32 each
    if (tid < 64) {
        int isD = tid >> 5;
        int q   = tid & 31;
        int s   = q >> 4, h = (q >> 2) & 3, f = q & 3;
        const float* W = s ? W_conf : W_coop;
        const float* a = isD ? (s ? ad_conf : ad_coop) : (s ? as_conf : as_coop);
        float acc = 0.f;
        #pragma unroll
        for (int o = 0; o < 32; o++) acc += W[h * 128 + f * 32 + o] * a[h * 32 + o];
        if (isD) wdsh[s][h][f] = acc; else wssh[s][h][f] = acc;
    }

    const int node = blockIdx.x * 4 + w;
    const int b    = node / 5;
    const int j    = node % 5;
    const float* src = (j == 0) ? (self_f + (size_t)b * 48)
                                : (nbr_f + ((size_t)b * 4 + (j - 1)) * 48);
    for (int e = lane; e < 48; e += 32) xsh[w][e] = src[e];

    for (int e = lane; e < 4 * AST; e += 32)              // zero pad rows 12..15
        Afrag[w][12 + e / AST][e % AST] = (_Float16)0.f;

    __syncthreads();

    #pragma unroll
    for (int s = 0; s < 2; s++) {
        const float* W = s ? W_conf : W_coop;
        #pragma unroll
        for (int h = 0; h < 4; h++) {
            // lanes 0..11: softmax row i, then y[i][:] = att_row @ x  (12x4)
            if (lane < 12) {
                int i = lane, gi = i / 3;
                float ws0 = wssh[s][h][0], ws1 = wssh[s][h][1], ws2 = wssh[s][h][2], ws3 = wssh[s][h][3];
                float wd0 = wdsh[s][h][0], wd1 = wdsh[s][h][1], wd2 = wdsh[s][h][2], wd3 = wdsh[s][h][3];
                const float* xi = &xsh[w][i * 4];
                float es = xi[0] * ws0 + xi[1] * ws1 + xi[2] * ws2 + xi[3] * ws3;
                float ev[12], mx = -3e38f;
                #pragma unroll
                for (int jj = 0; jj < 12; jj++) {
                    const float* xj = &xsh[w][jj * 4];
                    float ed = xj[0] * wd0 + xj[1] * wd1 + xj[2] * wd2 + xj[3] * wd3;
                    float e  = es + ed;
                    e = (e > 0.f) ? e : 0.2f * e;                     // leaky_relu(0.2)
                    bool allowed = (s == 0) ? ((jj / 3) == gi) : ((jj / 3) != gi);
                    ev[jj] = allowed ? e : -1e9f;
                    mx = fmaxf(mx, ev[jj]);
                }
                float sum = 0.f;
                #pragma unroll
                for (int jj = 0; jj < 12; jj++) { ev[jj] = __expf(ev[jj] - mx); sum += ev[jj]; }
                float inv = 1.f / sum;
                float y0 = 0.f, y1 = 0.f, y2 = 0.f, y3 = 0.f;
                #pragma unroll
                for (int jj = 0; jj < 12; jj++) {
                    float a = ev[jj] * inv;
                    const float* xj = &xsh[w][jj * 4];
                    y0 += a * xj[0]; y1 += a * xj[1]; y2 += a * xj[2]; y3 += a * xj[3];
                }
                ysh[w][i][0] = y0; ysh[w][i][1] = y1; ysh[w][i][2] = y2; ysh[w][i][3] = y3;
            }
            // all lanes (lane = o): out[i][o] = y[i][:] . W[h][:][o]; elu; -> Afrag
            float wc0 = W[h * 128 +  0 + lane];
            float wc1 = W[h * 128 + 32 + lane];
            float wc2 = W[h * 128 + 64 + lane];
            float wc3 = W[h * 128 + 96 + lane];
            #pragma unroll
            for (int i = 0; i < 12; i++) {
                float acc = ysh[w][i][0] * wc0 + ysh[w][i][1] * wc1
                          + ysh[w][i][2] * wc2 + ysh[w][i][3] * wc3;
                Afrag[w][i][s * 128 + h * 32 + lane] = (_Float16)elu_f(acc);
            }
        }
    }

    __syncthreads();

    // fuse GEMM: (16x256) @ (256x128), 8 k-steps x 8 n-tiles
    v8f acc[8] = {};
    const int m    = lane & 15;
    const int koff = (lane >> 4) * 16;
    const int bsel = (lane >> 4) << 4;
    #pragma unroll
    for (int kk = 0; kk < 8; kk++) {
        v16h a = load_a_frag(&Afrag[w][m][kk * 32 + koff]);
        #pragma unroll
        for (int t = 0; t < 8; t++) {
            v16h bb = load_b_frag(wfT + (size_t)(t * 16 + m) * 256 + kk * 32 + bsel);
            acc[t] = wmma_f16(a, bb, acc[t]);
        }
    }

    // +b_fuse, elu, mean over 12 rows (pad rows excluded via row count per lane-half)
    float* embn = emb_out + (size_t)node * EMBSZ;
    #pragma unroll
    for (int t = 0; t < 8; t++) {
        int   col  = t * 16 + m;
        float bf   = b_fuse[col];
        int   rows = (lane < 16) ? 8 : 4;
        float sme  = 0.f;
        #pragma unroll
        for (int i = 0; i < 8; i++)
            if (i < rows) sme += elu_f(acc[t][i] + bf);
        sme += __shfl_xor(sme, 16, 32);
        if (lane < 16) embn[col] = sme * (1.f / 12.f);
    }
}

// ---------------------------------------------------------------------------
// Kernel 2: head as WMMA. One wave = 16 batch rows (512 waves total).
//   proj (K=160 padded) -> relu*mask -> x_t ; 4-step GRU fwd+bwd ; out GEMM.
// ---------------------------------------------------------------------------
__global__ __launch_bounds__(64) void head_kernel(
    const float* __restrict__ emb,  const float* __restrict__ mask, const float* __restrict__ dirs,
    const _Float16* __restrict__ wt,
    const float* __restrict__ bp,
    const float* __restrict__ bihF, const float* __restrict__ bhhF,
    const float* __restrict__ bihB, const float* __restrict__ bhhB,
    const float* __restrict__ bo,   float* __restrict__ out)
{
    __shared__ __align__(16) _Float16 Abuf[2][16][ABST];
    __shared__ __align__(16) _Float16 xbuf[2][4][16][XST];
    __shared__ __align__(16) _Float16 hbuf[2][16][HST];

    const int tid  = threadIdx.x;
    const int w    = tid >> 5;
    const int lane = tid & 31;
    const int tile = blockIdx.x * 2 + w;
    const int b0   = tile * 16;
    const int m    = lane & 15;
    const int koff = (lane >> 4) * 16;
    const int bsel = (lane >> 4) << 4;

    const _Float16* WpT = wt + WPT_OFF;
    const _Float16* WoT = wt + WOT_OFF;

    // ---- projection, t = 0..3: x_t = relu([nbr_emb|dir] @ Wp + bp) * mask ----
    for (int t = 0; t < 4; t++) {
        for (int idx = lane; idx < 16 * 160; idx += 32) {
            int r = idx / 160, c = idx % 160;
            float v;
            if (c < 128)       v = emb[((size_t)(b0 + r) * 5 + 1 + t) * EMBSZ + c];
            else if (c == 128) v = dirs[(b0 + r) * 4 + t];
            else               v = 0.f;                       // K padding 129..159
            Abuf[w][r][c] = (_Float16)v;
        }
        float mk[8];
        #pragma unroll
        for (int i = 0; i < 8; i++)
            mk[i] = mask[(b0 + ((lane < 16) ? i : 8 + i)) * 4 + t];

        v8f pacc[4] = {};
        #pragma unroll
        for (int kk = 0; kk < 5; kk++) {
            v16h a = load_a_frag(&Abuf[w][m][kk * 32 + koff]);
            #pragma unroll
            for (int nt = 0; nt < 4; nt++) {
                v16h bb = load_b_frag(WpT + (size_t)(nt * 16 + m) * 160 + kk * 32 + bsel);
                pacc[nt] = wmma_f16(a, bb, pacc[nt]);
            }
        }
        #pragma unroll
        for (int nt = 0; nt < 4; nt++) {
            int   col = nt * 16 + m;
            float bpv = bp[col];
            #pragma unroll
            for (int i = 0; i < 8; i++) {
                int row = (lane < 16) ? i : 8 + i;
                xbuf[w][t][row][col] = (_Float16)(fmaxf(pacc[nt][i] + bpv, 0.f) * mk[i]);
            }
        }
    }

    // ---- bidirectional GRU, 4 steps each ----
    for (int d = 0; d < 2; d++) {
        const _Float16* WihT = wt + (d ? WIHB_OFF : WIHF_OFF);
        const _Float16* WhhT = wt + (d ? WHHB_OFF : WHHF_OFF);
        const float* bih = d ? bihB : bihF;
        const float* bhh = d ? bhhB : bhhF;

        float hreg[2][8];
        #pragma unroll
        for (int p = 0; p < 2; p++)
            #pragma unroll
            for (int i = 0; i < 8; i++) hreg[p][i] = 0.f;
        for (int idx = lane; idx < 16 * HST; idx += 32)
            hbuf[w][idx / HST][idx % HST] = (_Float16)0.f;

        for (int step = 0; step < 4; step++) {
            int t = d ? (3 - step) : step;
            v8f gi[6] = {}, gh[6] = {};
            #pragma unroll
            for (int kk = 0; kk < 2; kk++) {                 // gi = x_t @ Wih  (K=64)
                v16h a = load_a_frag(&xbuf[w][t][m][kk * 32 + koff]);
                #pragma unroll
                for (int nt = 0; nt < 6; nt++) {
                    v16h bb = load_b_frag(WihT + (size_t)(nt * 16 + m) * 64 + kk * 32 + bsel);
                    gi[nt] = wmma_f16(a, bb, gi[nt]);
                }
            }
            {                                                 // gh = h @ Whh  (K=32)
                v16h a = load_a_frag(&hbuf[w][m][koff]);
                #pragma unroll
                for (int nt = 0; nt < 6; nt++) {
                    v16h bb = load_b_frag(WhhT + (size_t)(nt * 16 + m) * 32 + bsel);
                    gh[nt] = wmma_f16(a, bb, gh[nt]);
                }
            }
            float biv[6], bhv[6];
            #pragma unroll
            for (int nt = 0; nt < 6; nt++) { int g = nt * 16 + m; biv[nt] = bih[g]; bhv[nt] = bhh[g]; }
            #pragma unroll
            for (int p = 0; p < 2; p++) {
                #pragma unroll
                for (int i = 0; i < 8; i++) {
                    float rv = sigm_f(gi[p][i]     + biv[p]     + gh[p][i]     + bhv[p]);
                    float zv = sigm_f(gi[2 + p][i] + biv[2 + p] + gh[2 + p][i] + bhv[2 + p]);
                    float nv = tanhf (gi[4 + p][i] + biv[4 + p] + rv * (gh[4 + p][i] + bhv[4 + p]));
                    float hn = (1.f - zv) * nv + zv * hreg[p][i];
                    hreg[p][i] = hn;
                    int row = (lane < 16) ? i : 8 + i;
                    hbuf[w][row][p * 16 + m] = (_Float16)hn;  // feeds next step's A fragment
                }
            }
        }
        // save final h into concat A (cols 128..159 fwd, 160..191 bwd)
        #pragma unroll
        for (int p = 0; p < 2; p++)
            #pragma unroll
            for (int i = 0; i < 8; i++) {
                int row = (lane < 16) ? i : 8 + i;
                Abuf[w][row][128 + d * 32 + p * 16 + m] = (_Float16)hreg[p][i];
            }
    }

    // ---- self_emb -> A cols 0..127 and out cols 0..127 ----
    for (int idx = lane; idx < 16 * 128; idx += 32) {
        int r = idx >> 7, c = idx & 127;
        float v = emb[((size_t)(b0 + r) * 5) * EMBSZ + c];
        Abuf[w][r][c] = (_Float16)v;
        out[(size_t)(b0 + r) * 192 + c] = v;
    }

    // ---- out GEMM: (16x192) @ (192x64), relu ----
    v8f oacc[4] = {};
    #pragma unroll
    for (int kk = 0; kk < 6; kk++) {
        v16h a = load_a_frag(&Abuf[w][m][kk * 32 + koff]);
        #pragma unroll
        for (int nt = 0; nt < 4; nt++) {
            v16h bb = load_b_frag(WoT + (size_t)(nt * 16 + m) * 192 + kk * 32 + bsel);
            oacc[nt] = wmma_f16(a, bb, oacc[nt]);
        }
    }
    #pragma unroll
    for (int nt = 0; nt < 4; nt++) {
        int   col = nt * 16 + m;
        float bov = bo[col];
        #pragma unroll
        for (int i = 0; i < 8; i++) {
            int row = (lane < 16) ? i : 8 + i;
            out[(size_t)(b0 + row) * 192 + 128 + col] = fmaxf(oacc[nt][i] + bov, 0.f);
        }
    }
}

// ---------------------------------------------------------------------------
extern "C" void kernel_launch(void* const* d_in, const int* in_sizes, int n_in,
                              void* d_out, int out_size, void* d_ws, size_t ws_size,
                              hipStream_t stream)
{
    const float* self_f  = (const float*)d_in[0];
    const float* nbr_f   = (const float*)d_in[1];
    const float* mask    = (const float*)d_in[2];
    const float* dirs    = (const float*)d_in[3];
    const float* W_coop  = (const float*)d_in[4];
    const float* as_coop = (const float*)d_in[5];
    const float* ad_coop = (const float*)d_in[6];
    const float* W_conf  = (const float*)d_in[7];
    const float* as_conf = (const float*)d_in[8];
    const float* ad_conf = (const float*)d_in[9];
    const float* W_fuse  = (const float*)d_in[10];
    const float* b_fuse  = (const float*)d_in[11];
    const float* Wp      = (const float*)d_in[12];
    const float* bp      = (const float*)d_in[13];
    const float* WihF    = (const float*)d_in[14];
    const float* WhhF    = (const float*)d_in[15];
    const float* bihF    = (const float*)d_in[16];
    const float* bhhF    = (const float*)d_in[17];
    const float* WihB    = (const float*)d_in[18];
    const float* WhhB    = (const float*)d_in[19];
    const float* bihB    = (const float*)d_in[20];
    const float* bhhB    = (const float*)d_in[21];
    const float* Wo      = (const float*)d_in[22];
    const float* bo      = (const float*)d_in[23];

    _Float16* wt  = (_Float16*)d_ws;                               // 147456 B arena
    float*    emb = (float*)((char*)d_ws + PREP_TOTAL * sizeof(_Float16));

    prep_weights<<<(PREP_TOTAL + 255) / 256, 256, 0, stream>>>(
        W_fuse, Wp, WihF, WihB, WhhF, WhhB, Wo, wt);

    gat_fuse_kernel<<<NODES / 4, 128, 0, stream>>>(
        self_f, nbr_f, W_coop, as_coop, ad_coop, W_conf, as_conf, ad_conf,
        wt + WFT_OFF, b_fuse, emb);

    head_kernel<<<(B_SZ / 16) / 2, 64, 0, stream>>>(
        emb, mask, dirs, wt, bp, bihF, bhhF, bihB, bhhB, bo, (float*)d_out);
}
